// TreeLSTM_38998303048339
// MI455X (gfx1250) — matrix operand order repaired and checked by reading, not compiled, not run on hardware
//
#include <hip/hip_runtime.h>
#include <hip/hip_bf16.h>
#include <stdint.h>

// ---------------- problem constants ----------------
#define B_   8
#define S_   2048
#define N_   4096          // nodes
#define H_   512           // hidden = OUT
#define M_   (B_ * S_)     // 16384 tokens
#define KD_  1536          // rel(512) + child0 h(512) + child1 h(512)
#define NG_  5             // gates: i, f0, f1, o, c
#define NT_  (NG_ * H_)    // 2560 packed output columns

typedef __bf16 bf16_t;
typedef __attribute__((ext_vector_type(16))) __bf16 v16bf;
typedef __attribute__((ext_vector_type(8)))  float  v8f;

union Frag { uint4 u[2]; v16bf v; };

__device__ __forceinline__ float fsigmoid(float x) { return 1.0f / (1.0f + __expf(-x)); }
__device__ __forceinline__ float ftanh(float x) {
    float e = __expf(-2.0f * fabsf(x));
    float t = (1.0f - e) / (1.0f + e);
    return copysignf(t, x);
}
__device__ __forceinline__ bf16_t f2bf(float f) { return (bf16_t)f; }

// ---------------- pack X: gather rel-emb + child hidden, f32 -> bf16 ----------------
__global__ void pack_x(const int* __restrict__ rid, const int* __restrict__ source,
                       const float* __restrict__ emb, const float* __restrict__ hidden,
                       bf16_t* __restrict__ X) {
    const int m = blockIdx.x;
    const int b = m >> 11;                      // S = 2048 = 2^11
    const float* rp = emb    + (size_t)rid[m] * H_;
    const float* h0 = hidden + ((size_t)b * N_ + source[2 * m    ]) * H_;
    const float* h1 = hidden + ((size_t)b * N_ + source[2 * m + 1]) * H_;
    bf16_t* xr = X + (size_t)m * KD_;
    for (int c = threadIdx.x; c < H_; c += 256) {
        xr[c]          = f2bf(rp[c]);
        xr[H_ + c]     = f2bf(h0[c]);
        xr[2 * H_ + c] = f2bf(h1[c]);
    }
}

// ---------------- pack W: (2560 x 1536) bf16, row n = gate g, out o ----------------
__global__ void pack_w(const float* __restrict__ Wi, const float* __restrict__ Ui,
                       const float* __restrict__ Wf, const float* __restrict__ Uf,
                       const float* __restrict__ Wo, const float* __restrict__ Uo,
                       const float* __restrict__ Wc, const float* __restrict__ Uc,
                       bf16_t* __restrict__ Wout) {
    const int n = blockIdx.x;                   // 0..2559
    const int g = n >> 9, o = n & 511;
    const float* wp; const float* up;
    switch (g) {
        case 0:  wp = Wi + (size_t)o * H_;          up = Ui + (size_t)o * 2 * H_;          break;
        case 1:  wp = Wf + (size_t)o * H_;          up = Uf + (size_t)o * 2 * H_;          break;
        case 2:  wp = Wf + (size_t)(512 + o) * H_;  up = Uf + (size_t)(512 + o) * 2 * H_;  break;
        case 3:  wp = Wo + (size_t)o * H_;          up = Uo + (size_t)o * 2 * H_;          break;
        default: wp = Wc + (size_t)o * H_;          up = Uc + (size_t)o * 2 * H_;          break;
    }
    bf16_t* dst = Wout + (size_t)n * KD_;
    for (int c = threadIdx.x; c < H_; c += 256)     dst[c]      = f2bf(wp[c]);
    for (int c = threadIdx.x; c < 2 * H_; c += 256) dst[H_ + c] = f2bf(up[c]);
}

__global__ void pack_bias(const float* __restrict__ Wib, const float* __restrict__ Uib,
                          const float* __restrict__ Wfb, const float* __restrict__ Ufb,
                          const float* __restrict__ Wob, const float* __restrict__ Uob,
                          const float* __restrict__ Wcb, const float* __restrict__ Ucb,
                          float* __restrict__ bias) {
    const int n = blockIdx.x * 256 + threadIdx.x;
    if (n >= NT_) return;
    const int g = n >> 9, o = n & 511;
    float v;
    switch (g) {
        case 0:  v = Wib[o] + Uib[o];             break;
        case 1:  v = Wfb[o] + Ufb[o];             break;
        case 2:  v = Wfb[512 + o] + Ufb[512 + o]; break;
        case 3:  v = Wob[o] + Uob[o];             break;
        default: v = Wcb[o] + Ucb[o];             break;
    }
    bias[n] = v;
}

// ---------------- fused GEMM (WMMA bf16 -> f32) + LSTM epilogue ----------------
// grid = (M/128, H/64); block = 256 threads = 8 waves.
// wave (wm in 0..3, wo in 0..1): 32 M-rows x 32 O-cols x 5 gates = 20 accum tiles.
// Weight slab (320 rows x 32 K) is double-buffered in LDS and filled with the
// CDNA5 async-to-LDS engine (ASYNCcnt), overlapped with the 20 WMMAs per step.
#define MT_       128
#define KSTEP     32
#define NSTEPS    (KD_ / KSTEP)     // 48
#define LDSROWS   320               // 5 gates * 64 o-cols
#define LDSSTRIDE 40                // bf16 elems per row (80 B, padded)
#define LDSBUF    (LDSROWS * LDSSTRIDE)

__global__ __launch_bounds__(256)
void treelstm_wmma(const bf16_t* __restrict__ X, const bf16_t* __restrict__ W,
                   const float* __restrict__ bias, const int* __restrict__ source,
                   const float* __restrict__ ctx,
                   float* __restrict__ outH, float* __restrict__ outC) {
    __shared__ __align__(16) bf16_t ldsB[2][LDSBUF];

    const int tid  = threadIdx.x;
    const int lane = tid & 31;
    const int wave = tid >> 5;
    const int half = lane >> 4;     // K half-select for fragments
    const int l16  = lane & 15;
    const int wm   = wave & 3;      // M sub-tile (32 rows each)
    const int wo   = wave >> 2;     // O sub-tile (32 cols each)

    const int mBlk = (int)blockIdx.x * MT_;
    const int oBlk = (int)blockIdx.y * 64;

    v8f acc[NG_][2][2];             // [gate][mt][t]
    const v8f vzero = {0.f, 0.f, 0.f, 0.f, 0.f, 0.f, 0.f, 0.f};
#pragma unroll
    for (int g = 0; g < NG_; ++g)
#pragma unroll
        for (int mt = 0; mt < 2; ++mt)
#pragma unroll
            for (int t = 0; t < 2; ++t) acc[g][mt][t] = vzero;

    // per-thread async-fill descriptors: 1280 16-B chunks, 5 per thread
    const bf16_t* gsrc[5];
    uint32_t      loff[5];          // LDS byte offset within buffer 0
#pragma unroll
    for (int ii = 0; ii < 5; ++ii) {
        const int idx  = tid + ii * 256;
        const int row  = idx >> 2;          // 0..319
        const int part = idx & 3;           // 8-element chunk
        const int g    = row >> 6;
        const int ol   = row & 63;
        gsrc[ii] = W + (size_t)(g * H_ + oBlk + ol) * KD_ + part * 8;
        loff[ii] = (uint32_t)(uintptr_t)&ldsB[0][row * LDSSTRIDE + part * 8];
    }

    // issue the async fill of one 320x32 weight slab into buffer `buf`
    auto fill = [&](int kk, int buf) {
#pragma unroll
        for (int ii = 0; ii < 5; ++ii) {
            const uint32_t la = loff[ii] + (uint32_t)buf * (uint32_t)(LDSBUF * sizeof(bf16_t));
            const unsigned long long ga = (unsigned long long)(uintptr_t)(gsrc[ii] + kk);
            asm volatile("global_load_async_to_lds_b128 %0, %1, off"
                         :: "v"(la), "v"(ga) : "memory");
        }
    };

    // A fragment bases: ISA 16-bit A 16x32 layout -> lane(half,m) reads
    // K chunks [k0, k0+8) and [k0+16, k0+24) with k0 = kk + half*8.
    const bf16_t* aBase0 = X + (size_t)(mBlk + wm * 32 + l16) * KD_ + half * 8;
    const bf16_t* aBase1 = aBase0 + (size_t)16 * KD_;

    // prologue: fill buffer 0, wait, publish
    fill(0, 0);
    asm volatile("s_wait_asynccnt 0" ::: "memory");
    __syncthreads();

    for (int s = 0; s < NSTEPS; ++s) {
        const int kk  = s * KSTEP;
        const int cur = s & 1;
        if (s + 1 < NSTEPS) fill(kk + KSTEP, cur ^ 1);   // overlap with compute

        Frag a0, a1;
        a0.u[0] = *(const uint4*)(aBase0 + kk);
        a0.u[1] = *(const uint4*)(aBase0 + kk + 16);
        a1.u[0] = *(const uint4*)(aBase1 + kk);
        a1.u[1] = *(const uint4*)(aBase1 + kk + 16);

        const bf16_t* lb = ldsB[cur];
#pragma unroll
        for (int g = 0; g < NG_; ++g) {
#pragma unroll
            for (int t = 0; t < 2; ++t) {
                // B fragment (ISA dense-B layout): lane = column N,
                // elements are K chunks [half*8, half*8+8) and [16+half*8, 24+half*8).
                const int nl = g * 64 + wo * 32 + t * 16 + l16;
                const bf16_t* bp = lb + nl * LDSSTRIDE + half * 8;
                Frag bfr;
                bfr.u[0] = *(const uint4*)bp;
                bfr.u[1] = *(const uint4*)(bp + 16);
                acc[g][0][t] = __builtin_amdgcn_wmma_f32_16x16x32_bf16(
                    false, a0.v, false, bfr.v, (short)0, acc[g][0][t], false, false);
                acc[g][1][t] = __builtin_amdgcn_wmma_f32_16x16x32_bf16(
                    false, a1.v, false, bfr.v, (short)0, acc[g][1][t], false, false);
            }
        }

        asm volatile("s_wait_asynccnt 0" ::: "memory");  // next buffer landed
        __syncthreads();
    }

    // ---------------- epilogue: bias + gates + context gather ----------------
    // C/D layout: VGPR j -> M = j + half*8 ; N = l16.
#pragma unroll
    for (int mt = 0; mt < 2; ++mt) {
#pragma unroll
        for (int j = 0; j < 8; ++j) {
            const int m  = mBlk + wm * 32 + mt * 16 + half * 8 + j;
            const int b  = m >> 11;
            const int i0 = source[2 * m];
            const int i1 = source[2 * m + 1];
            const float* c0 = ctx + ((size_t)b * N_ + i0) * H_;
            const float* c1 = ctx + ((size_t)b * N_ + i1) * H_;
#pragma unroll
            for (int t = 0; t < 2; ++t) {
                const int o = oBlk + wo * 32 + t * 16 + l16;
                const float pi  = acc[0][mt][t][j] + bias[o];
                const float pf0 = acc[1][mt][t][j] + bias[512 + o];
                const float pf1 = acc[2][mt][t][j] + bias[1024 + o];
                const float po  = acc[3][mt][t][j] + bias[1536 + o];
                const float pc  = acc[4][mt][t][j] + bias[2048 + o];
                const float ig = fsigmoid(pi);
                const float f0 = fsigmoid(pf0);
                const float f1 = fsigmoid(pf1);
                const float og = fsigmoid(po);
                const float cg = ftanh(pc);
                const float cv = ig * cg + f0 * c0[o] + f1 * c1[o];
                const float hv = ftanh(cv) * og;
                const size_t off = (size_t)m * H_ + o;
                outH[off] = hv;
                outC[off] = cv;
            }
        }
    }
}

// ---------------- launch ----------------
extern "C" void kernel_launch(void* const* d_in, const int* in_sizes, int n_in,
                              void* d_out, int out_size, void* d_ws, size_t ws_size,
                              hipStream_t stream) {
    const int*   rid  = (const int*)  d_in[0];
    const float* hid  = (const float*)d_in[1];
    const float* ctx  = (const float*)d_in[2];
    const int*   src  = (const int*)  d_in[3];
    const float* emb  = (const float*)d_in[4];
    const float* Wi_w = (const float*)d_in[5];
    const float* Wi_b = (const float*)d_in[6];
    const float* Ui_w = (const float*)d_in[7];
    const float* Ui_b = (const float*)d_in[8];
    const float* Wf_w = (const float*)d_in[9];
    const float* Wf_b = (const float*)d_in[10];
    const float* Uf_w = (const float*)d_in[11];
    const float* Uf_b = (const float*)d_in[12];
    const float* Wo_w = (const float*)d_in[13];
    const float* Wo_b = (const float*)d_in[14];
    const float* Uo_w = (const float*)d_in[15];
    const float* Uo_b = (const float*)d_in[16];
    const float* Wc_w = (const float*)d_in[17];
    const float* Wc_b = (const float*)d_in[18];
    const float* Uc_w = (const float*)d_in[19];
    const float* Uc_b = (const float*)d_in[20];

    // workspace layout (all 256-B aligned)
    const size_t xBytes = (size_t)M_ * KD_ * sizeof(bf16_t);   // 50,331,648
    const size_t wBytes = (size_t)NT_ * KD_ * sizeof(bf16_t);  //  7,864,320
    char* ws = (char*)d_ws;
    bf16_t* X    = (bf16_t*)ws;
    bf16_t* Wcat = (bf16_t*)(ws + xBytes);
    float*  bias = (float*) (ws + xBytes + wBytes);

    float* outH = (float*)d_out;
    float* outC = outH + (size_t)M_ * H_;

    pack_x   <<<dim3(M_),  dim3(256), 0, stream>>>(rid, src, emb, hid, X);
    pack_w   <<<dim3(NT_), dim3(256), 0, stream>>>(Wi_w, Ui_w, Wf_w, Uf_w, Wo_w, Uo_w, Wc_w, Uc_w, Wcat);
    pack_bias<<<dim3(10),  dim3(256), 0, stream>>>(Wi_b, Ui_b, Wf_b, Uf_b, Wo_b, Uo_b, Wc_b, Uc_b, bias);
    treelstm_wmma<<<dim3(M_ / MT_, H_ / 64), dim3(256), 0, stream>>>(X, Wcat, bias, src, ctx, outH, outC);
}
